// CausalSelfAttention_38843684225620
// MI455X (gfx1250) — matrix operand order repaired and checked by reading, not compiled, not run on hardware
//
#include <hip/hip_runtime.h>

// Problem constants (from reference)
#define B_    4
#define T_    2048
#define TP_   256
#define TALL_ 2304          // T_P + T
#define C_    768
#define H_    12
#define HD_   64

typedef __attribute__((ext_vector_type(16))) _Float16 v16h;
typedef __attribute__((ext_vector_type(8)))  float    v8f;

union Frag16 { v16h v; uint4 u[2]; };

// ---------------------------------------------------------------------------
// WMMA fragment loaders (v_wmma_f32_16x16x32_f16 operands, wave32; K-contiguous order used here, same for A and B)
//
// A (16x32, M x K), source row-major (M rows, leading dim ld):
//   lanes 0-15 : M = lane,    halves 0-15 = K[0..15]
//   lanes 16-31: M = lane-16, halves 0-15 = K[16..31]
// -> two b128 loads per lane, K-contiguous.
//
// B (32x16, K x N), source stored (N,K) ROW-MAJOR (i.e. B[k][n] = src[n][k]):
//   lanes 0-15 : N = lane,    halves 0-15 = K[0..15]
//   lanes 16-31: N = lane-16, halves 0-15 = K[16..31]
// -> two b128 loads per lane, K-contiguous.
// ---------------------------------------------------------------------------
__device__ __forceinline__ v16h load_a_frag(const _Float16* src, int ld,
                                            int row0, int kbase, int lane) {
  const int l = lane & 15, hi = (lane >> 4) & 1;
  const _Float16* p = src + (size_t)(row0 + l) * ld + (kbase + hi * 16);
  Frag16 f;
  f.u[0] = *(const uint4*)(p);        // K +0..7
  f.u[1] = *(const uint4*)(p + 8);    // K +8..15
  return f.v;
}

__device__ __forceinline__ v16h load_b_frag(const _Float16* src, int ld,
                                            int n0, int kbase, int lane) {
  const int l = lane & 15, hi = (lane >> 4) & 1;
  const _Float16* p = src + (size_t)(n0 + l) * ld + (kbase + hi * 16);
  Frag16 f;
  f.u[0] = *(const uint4*)(p);        // K +0..7
  f.u[1] = *(const uint4*)(p + 8);    // K +8..15
  return f.v;
}

__device__ __forceinline__ v8f wmma16(v16h a, v16h b, v8f c) {
  return __builtin_amdgcn_wmma_f32_16x16x32_f16(false, a, false, b,
                                                (short)0, c, false, false);
}

// ---------------------------------------------------------------------------
// fp32 -> f16 conversion kernels
// ---------------------------------------------------------------------------
__global__ void cvt_concat_kernel(const float* __restrict__ x,
                                  const float* __restrict__ prefix,
                                  _Float16* __restrict__ xh, int n) {
  int i = blockIdx.x * blockDim.x + threadIdx.x;
  if (i >= n) return;
  int c = i % C_;
  int t = (i / C_) % TALL_;
  int b = i / (C_ * TALL_);
  float v = (t < TP_) ? prefix[((size_t)b * TP_ + t) * C_ + c]
                      : x[((size_t)b * T_ + (t - TP_)) * C_ + c];
  xh[i] = (_Float16)v;
}

__global__ void cvt_f16_kernel(const float* __restrict__ src,
                               _Float16* __restrict__ dst, int n) {
  int i = blockIdx.x * blockDim.x + threadIdx.x;
  if (i < n) dst[i] = (_Float16)src[i];
}

// ---------------------------------------------------------------------------
// QKV projection: (B*T_all, C) x W_qkv^T -> Q (B,H,Tall,HD), K (B,H,Tall,HD),
// V stored TRANSPOSED as Vt (B,H,HD,Tall) so P@V can use the (N,K) B-loader.
// Per-wave 32x32 output tile, K loop over 768 in steps of 32.
// ---------------------------------------------------------------------------
__global__ __launch_bounds__(256) void qkv_gemm_kernel(
    const _Float16* __restrict__ xh, const _Float16* __restrict__ wq,
    _Float16* __restrict__ qb, _Float16* __restrict__ kb,
    _Float16* __restrict__ vt) {
  const int lane = threadIdx.x & 31;
  const int tile = blockIdx.x * 8 + (threadIdx.x >> 5);
  const int NT = (3 * C_) / 32;                 // 72 col tiles
  const int nt = tile % NT, mt = tile / NT;     // mt < 288
  const int row0 = mt * 32, col0 = nt * 32;

  v8f acc[2][2] = {};
  for (int k = 0; k < C_; k += 32) {
    v16h a0 = load_a_frag(xh, C_, row0,      k, lane);
    v16h a1 = load_a_frag(xh, C_, row0 + 16, k, lane);
    v16h b0 = load_b_frag(wq, C_, col0,      k, lane);
    v16h b1 = load_b_frag(wq, C_, col0 + 16, k, lane);
    acc[0][0] = wmma16(a0, b0, acc[0][0]);
    acc[0][1] = wmma16(a0, b1, acc[0][1]);
    acc[1][0] = wmma16(a1, b0, acc[1][0]);
    acc[1][1] = wmma16(a1, b1, acc[1][1]);
  }

  const int hi = (lane >> 4) & 1, l = lane & 15;
  #pragma unroll
  for (int i = 0; i < 2; i++)
    #pragma unroll
    for (int j = 0; j < 2; j++)
      #pragma unroll
      for (int v = 0; v < 8; v++) {
        int M = row0 + i * 16 + v + 8 * hi;   // token row (C layout)
        int N = col0 + j * 16 + l;            // output feature
        int b   = M / TALL_, tok = M % TALL_;
        int whi = N / C_,    c   = N % C_;
        int h = c >> 6, d = c & 63;
        _Float16 val = (_Float16)acc[i][j][v];
        size_t bh = (size_t)(b * H_ + h);
        if (whi == 0)      qb[(bh * TALL_ + tok) * HD_ + d] = val;
        else if (whi == 1) kb[(bh * TALL_ + tok) * HD_ + d] = val;
        else               vt[(bh * HD_ + d) * TALL_ + tok] = val;
      }
}

// ---------------------------------------------------------------------------
// Flash attention, one 16-row q-tile per wave, q restricted to x rows
// (plain causal in global index). 32-key blocks: 4 WMMAs for S, online
// softmax with shfl_xor row reductions, P through wave-private LDS, 4 WMMAs
// for P @ V^T.
// ---------------------------------------------------------------------------
__global__ __launch_bounds__(256) void attn_kernel(
    const _Float16* __restrict__ qb, const _Float16* __restrict__ kb,
    const _Float16* __restrict__ vt, _Float16* __restrict__ oh) {
  __shared__ _Float16 pshare[8][16 * 32];     // one 16x32 P tile per wave
  const int lane = threadIdx.x & 31;
  const int w = threadIdx.x >> 5;
  const int id = blockIdx.x * 8 + w;
  const int qt = id & 127;                    // 128 q-tiles of 16 rows
  const int h  = (id >> 7) % H_;
  const int b  = id / (128 * H_);
  const size_t bh = (size_t)(b * H_ + h);
  const _Float16* Q  = qb + bh * TALL_ * HD_;
  const _Float16* K  = kb + bh * TALL_ * HD_;
  const _Float16* Vt = vt + bh * HD_ * TALL_;
  const int qbase = TP_ + qt * 16;            // global query row base

  // Q fragments, loaded once (HD=64 -> two K-steps)
  v16h qf0 = load_a_frag(Q, HD_, qbase, 0,  lane);
  v16h qf1 = load_a_frag(Q, HD_, qbase, 32, lane);

  v8f o[4] = {};                              // O accumulator, 4 N-tiles of HD
  float mrow[8], lrow[8];
  #pragma unroll
  for (int v = 0; v < 8; v++) { mrow[v] = -1e30f; lrow[v] = 0.f; }

  const int hi = (lane >> 4) & 1, l = lane & 15;
  const int nb = (qbase + 16 + 31) >> 5;      // 32-key blocks needed
  _Float16* pl = pshare[w];

  for (int kbk = 0; kbk < nb; kbk++) {
    const int k0 = kbk * 32;
    // S = Q @ K^T for 32 keys (two 16-col subtiles)
    v8f s0 = {}, s1 = {};
    s0 = wmma16(qf0, load_b_frag(K, HD_, k0,      0,  lane), s0);
    s0 = wmma16(qf1, load_b_frag(K, HD_, k0,      32, lane), s0);
    s1 = wmma16(qf0, load_b_frag(K, HD_, k0 + 16, 0,  lane), s1);
    s1 = wmma16(qf1, load_b_frag(K, HD_, k0 + 16, 32, lane), s1);

    const int key0 = k0 + l, key1 = k0 + 16 + l;
    float mnew[8];
    #pragma unroll
    for (int v = 0; v < 8; v++) {
      const int q = qbase + v + 8 * hi;       // global query row for this slot
      float a = s0[v] * 0.125f; if (key0 > q) a = -1e30f;
      float c = s1[v] * 0.125f; if (key1 > q) c = -1e30f;
      s0[v] = a; s1[v] = c;
      float mv = fmaxf(a, c);                 // reduce across the 16-lane half
      mv = fmaxf(mv, __shfl_xor(mv, 1));
      mv = fmaxf(mv, __shfl_xor(mv, 2));
      mv = fmaxf(mv, __shfl_xor(mv, 4));
      mv = fmaxf(mv, __shfl_xor(mv, 8));
      mnew[v] = fmaxf(mrow[v], mv);
    }
    #pragma unroll
    for (int v = 0; v < 8; v++) {
      float p0 = __expf(s0[v] - mnew[v]);
      float p1 = __expf(s1[v] - mnew[v]);
      float alpha = __expf(mrow[v] - mnew[v]);
      float rs = p0 + p1;
      rs += __shfl_xor(rs, 1); rs += __shfl_xor(rs, 2);
      rs += __shfl_xor(rs, 4); rs += __shfl_xor(rs, 8);
      lrow[v] = lrow[v] * alpha + rs;
      mrow[v] = mnew[v];
      #pragma unroll
      for (int j = 0; j < 4; j++) o[j][v] *= alpha;
      // C-layout (row = v+8*hi, col = l / 16+l) -> row-major P tile in LDS
      pl[(v + 8 * hi) * 32 + l]      = (_Float16)p0;
      pl[(v + 8 * hi) * 32 + 16 + l] = (_Float16)p1;
    }
    // same-wave DS ops are in-order; just drain DScnt before re-reading
    asm volatile("s_wait_dscnt 0" ::: "memory");
    v16h pf = load_a_frag(pl, 32, 0, 0, lane);     // P as A fragment (16x32)
    #pragma unroll
    for (int j = 0; j < 4; j++)
      o[j] = wmma16(pf, load_b_frag(Vt, TALL_, j * 16, k0, lane), o[j]);
  }

  // epilogue: O /= l, store f16 to attn buffer (B, T, C) [x rows only]
  #pragma unroll
  for (int v = 0; v < 8; v++) {
    float inv = 1.0f / lrow[v];
    int tloc = qt * 16 + v + 8 * hi;               // local x-row index
    size_t base = ((size_t)b * T_ + tloc) * C_ + h * HD_;
    #pragma unroll
    for (int j = 0; j < 4; j++)
      oh[base + j * 16 + l] = (_Float16)(o[j][v] * inv);
  }
}

// ---------------------------------------------------------------------------
// Output projection: attn (B*T, C) f16 x W_out^T -> out fp32 (B,T,C)
// ---------------------------------------------------------------------------
__global__ __launch_bounds__(256) void out_gemm_kernel(
    const _Float16* __restrict__ ah, const _Float16* __restrict__ wo,
    float* __restrict__ out) {
  const int lane = threadIdx.x & 31;
  const int tile = blockIdx.x * 8 + (threadIdx.x >> 5);
  const int NT = C_ / 32;                       // 24 col tiles
  const int nt = tile % NT, mt = tile / NT;     // mt < 256
  const int row0 = mt * 32, col0 = nt * 32;

  v8f acc[2][2] = {};
  for (int k = 0; k < C_; k += 32) {
    v16h a0 = load_a_frag(ah, C_, row0,      k, lane);
    v16h a1 = load_a_frag(ah, C_, row0 + 16, k, lane);
    v16h b0 = load_b_frag(wo, C_, col0,      k, lane);
    v16h b1 = load_b_frag(wo, C_, col0 + 16, k, lane);
    acc[0][0] = wmma16(a0, b0, acc[0][0]);
    acc[0][1] = wmma16(a0, b1, acc[0][1]);
    acc[1][0] = wmma16(a1, b0, acc[1][0]);
    acc[1][1] = wmma16(a1, b1, acc[1][1]);
  }

  const int hi = (lane >> 4) & 1, l = lane & 15;
  #pragma unroll
  for (int i = 0; i < 2; i++)
    #pragma unroll
    for (int j = 0; j < 2; j++)
      #pragma unroll
      for (int v = 0; v < 8; v++) {
        int M = row0 + i * 16 + v + 8 * hi;
        int N = col0 + j * 16 + l;
        out[(size_t)M * C_ + N] = acc[i][j][v];
      }
}

// ---------------------------------------------------------------------------
// Host launcher.  Workspace layout (f16 elements), total ~74 MB:
//   xh     : B*Tall*C   = 7,077,888
//   wqkvh  : 3*C*C      = 1,769,472
//   wouth  : C*C        =   589,824
//   qb     : B*H*Tall*HD= 7,077,888
//   kb     : B*H*Tall*HD= 7,077,888
//   vtb    : B*H*HD*Tall= 7,077,888
//   ah     : B*T*C      = 6,291,456
// ---------------------------------------------------------------------------
extern "C" void kernel_launch(void* const* d_in, const int* in_sizes, int n_in,
                              void* d_out, int out_size, void* d_ws, size_t ws_size,
                              hipStream_t stream) {
  const float* x      = (const float*)d_in[0];
  const float* prefix = (const float*)d_in[1];
  const float* wqkv   = (const float*)d_in[2];
  const float* wout   = (const float*)d_in[3];
  float* out = (float*)d_out;

  _Float16* xh    = (_Float16*)d_ws;
  _Float16* wqkvh = xh    + (size_t)B_ * TALL_ * C_;
  _Float16* wouth = wqkvh + (size_t)3 * C_ * C_;
  _Float16* qb    = wouth + (size_t)C_ * C_;
  _Float16* kb    = qb    + (size_t)B_ * H_ * TALL_ * HD_;
  _Float16* vtb   = kb    + (size_t)B_ * H_ * TALL_ * HD_;
  _Float16* ah    = vtb   + (size_t)B_ * H_ * HD_ * TALL_;

  const int n_xfull = B_ * TALL_ * C_;
  cvt_concat_kernel<<<n_xfull / 256, 256, 0, stream>>>(x, prefix, xh, n_xfull);
  cvt_f16_kernel<<<(3 * C_ * C_) / 256, 256, 0, stream>>>(wqkv, wqkvh, 3 * C_ * C_);
  cvt_f16_kernel<<<(C_ * C_) / 256, 256, 0, stream>>>(wout, wouth, C_ * C_);

  // QKV GEMM: (9216/32) x (2304/32) = 20736 wave tiles / 8 waves per block
  qkv_gemm_kernel<<<20736 / 8, 256, 0, stream>>>(xh, wqkvh, qb, kb, vtb);

  // Attention: B*H*(T/16) = 6144 wave tiles / 8 waves per block
  attn_kernel<<<6144 / 8, 256, 0, stream>>>(qb, kb, vtb, ah);

  // Output projection: (8192/32) x (768/32) = 6144 wave tiles / 8
  out_gemm_kernel<<<6144 / 8, 256, 0, stream>>>(ah, wouth, out);
}
